// ChromaSelfAttention_9732395893180
// MI455X (gfx1250) — compile-verified, matrix operand order checked by reading, not run on hardware
//
#include <hip/hip_runtime.h>
#include <hip/hip_bf16.h>

// ---------------------------------------------------------------------------
// ChromaSelfAttention on gfx1250 (MI455X): bf16 WMMA GEMMs + flash attention.
// B=2, L=2048, D=2048, H=16, HD=128.
//  - all WMMA fragment reads are contiguous-per-lane (ds_load_b128 pairs)
//  - global->LDS staging uses CDNA5 async-DMA (global_load_async_to_lds_b128,
//    ASYNCcnt) with double-buffered LDS tiles: DMA of tile i+1 overlaps the
//    WMMAs of tile i, one workgroup barrier per tile.
//  - attention computes S^T = K Q^T and O^T = V^T P^T so P stays in registers
//    and V (written transposed by its projection) is read contiguously.
// ---------------------------------------------------------------------------

typedef __bf16 bf16;
typedef __attribute__((ext_vector_type(16))) __bf16 bf16x16;
typedef __attribute__((ext_vector_type(8)))  float  floatx8;

#define NUM_HEADS 16
#define HEAD_DIM  128
#define SEQ_LEN   2048
#define MODEL_DIM 2048
#define BATCH     2
#define M_TOTAL   (BATCH * SEQ_LEN)          // 4096
#define ATTN_SCALE 0.08838834764831845f      // 128^-0.5

// K-index mapping for 16-bit WMMA A/B fragments (CDNA5 ISA 7.12.2):
// lane in [0,16): b0 = 0 ; lane in [16,32): b0 = 8
// element e in v16 -> K = b0 + e + (e>=8 ? 8 : 0)  (two 8-half contiguous runs)
__device__ __forceinline__ int kmap(int e, int b0) {
    return b0 + e + ((e >= 8) ? 8 : 0);
}

__device__ __forceinline__ floatx8 wmma_bf16(bf16x16 a, bf16x16 b, floatx8 c) {
    return __builtin_amdgcn_wmma_f32_16x16x32_bf16(false, a, false, b,
                                                   (short)0, c, false, false);
}

// ---------------------------------------------------------------------------
// CDNA5 async global -> LDS copy (16B per lane, tracked by ASYNCcnt).
// Generic pointers to __shared__ carry the wave-relative LDS byte offset in
// their low 32 bits (ISA 10.2 aperture rules), which is what VDST wants.
// ---------------------------------------------------------------------------
__device__ __forceinline__ void async_copy_b128(void* lds_ptr, const void* gptr) {
    unsigned int loff = (unsigned int)(uintptr_t)lds_ptr;
    unsigned long long gaddr = (unsigned long long)(uintptr_t)gptr;
    asm volatile("global_load_async_to_lds_b128 %0, %1, off"
                 :: "v"(loff), "v"(gaddr)
                 : "memory");
}

__device__ __forceinline__ void wait_async_all() {
    asm volatile("s_wait_asynccnt 0x0" ::: "memory");
}

// ---------------------------------------------------------------------------
// f32 -> bf16 conversion (activations, row-major preserved)
// ---------------------------------------------------------------------------
__global__ __launch_bounds__(256)
void cvt_f32_to_bf16(const float* __restrict__ src, bf16* __restrict__ dst, int n) {
    int i = (blockIdx.x * 256 + threadIdx.x) * 4;
    if (i + 3 < n) {
        float4 v = *(const float4*)(src + i);
        dst[i + 0] = (bf16)v.x;
        dst[i + 1] = (bf16)v.y;
        dst[i + 2] = (bf16)v.z;
        dst[i + 3] = (bf16)v.w;
    }
}

// ---------------------------------------------------------------------------
// f32 -> bf16 + transpose: WT[n][k] = W[k][n].  LDS-tiled 32x32, both sides
// coalesced.  One-time cost; weights are reused by 32 M-blocks afterwards.
// ---------------------------------------------------------------------------
__global__ __launch_bounds__(256)
void cvt_transpose_f32_bf16(const float* __restrict__ W, bf16* __restrict__ WT,
                            int N) {
    __shared__ float tile[32][33];
    const int t  = threadIdx.x;
    const int bn = blockIdx.x * 32;   // n base (output row)
    const int bk = blockIdx.y * 32;   // k base (output col)
    const int r  = t >> 3, c = (t & 7) * 4;

    float4 v = *(const float4*)(W + (size_t)(bk + r) * N + bn + c);
    tile[r][c + 0] = v.x; tile[r][c + 1] = v.y;
    tile[r][c + 2] = v.z; tile[r][c + 3] = v.w;
    __syncthreads();

    union { bf16 h[4]; uint2 u; } pk;
#pragma unroll
    for (int i = 0; i < 4; ++i) pk.h[i] = (bf16)tile[c + i][r];
    *(uint2*)(WT + (size_t)(bn + r) * N + bk + c) = pk.u;
}

// ---------------------------------------------------------------------------
// bf16 WMMA GEMM: C[M,N] = A[M,K] @ W[K,N] + bias, W given TRANSPOSED (WT[N][K]).
// Block tile: 128(M) x 64(N); 8 waves, each wave = 16 rows x 4 n-subtiles.
// Double-buffered async LDS staging; one barrier per 32-wide K chunk.
// mode 0: bf16 split-head [bh][l][d]            (Q, K projections)
// mode 2: bf16 split-head transposed [bh][d][l] (V projection)
// mode 1: f32 row-major                         (final projection -> d_out)
// ---------------------------------------------------------------------------
__global__ __launch_bounds__(256)
void gemm_bf16_wmma(const bf16* __restrict__ A, const bf16* __restrict__ WT,
                    const float* __restrict__ bias,
                    bf16* __restrict__ outB, float* __restrict__ outF,
                    int M, int N, int K, int mode) {
    const int t    = threadIdx.x;
    const int lane = t & 31;
    const int wave = t >> 5;
    const int n0   = blockIdx.x * 64;
    const int m0   = blockIdx.y * 128;
    const int b0   = (lane >= 16) ? 8 : 0;
    const int n16  = lane & 15;

    // row strides in halves: 40*2=80B (16B multiple) -> b128-friendly
    __shared__ bf16 As [2][128 * 40];   // A tile, [m][k]
    __shared__ bf16 BsT[2][ 64 * 40];   // W tile, [n][k] (already transposed)

    // per-thread staging coordinates (A: 2 threads/row, B: 4 threads/row)
    const int ar = t >> 1, ac = (t & 1) * 16;
    const int br = t >> 2, bc = (t & 3) * 8;
    const bf16* Agl = A  + (size_t)(m0 + ar) * K + ac;
    const bf16* Bgl = WT + (size_t)(n0 + br) * K + bc;
    const int aoff = ar * 40 + ac;
    const int boff = br * 40 + bc;

    floatx8 acc[4] = {};

    // prologue: async-stage tile 0
    async_copy_b128(&As [0][aoff],     Agl);
    async_copy_b128(&As [0][aoff + 8], Agl + 8);
    async_copy_b128(&BsT[0][boff],     Bgl);

    int pb = 0;
    for (int kc = 0; kc < K; kc += 32) {
        wait_async_all();
        __syncthreads();

        if (kc + 32 < K) {   // kick off DMA for next tile into alternate buffer
            int nb = pb ^ 1;
            async_copy_b128(&As [nb][aoff],     Agl + kc + 32);
            async_copy_b128(&As [nb][aoff + 8], Agl + kc + 40);
            async_copy_b128(&BsT[nb][boff],     Bgl + kc + 32);
        }

        bf16x16 a;
#pragma unroll
        for (int e = 0; e < 16; ++e)
            a[e] = As[pb][(wave * 16 + n16) * 40 + kmap(e, b0)];

#pragma unroll
        for (int nn = 0; nn < 4; ++nn) {
            bf16x16 bf;
#pragma unroll
            for (int e = 0; e < 16; ++e)
                bf[e] = BsT[pb][(nn * 16 + n16) * 40 + kmap(e, b0)];
            acc[nn] = wmma_bf16(a, bf, acc[nn]);
        }
        pb ^= 1;
    }

    // epilogue: C row = r + b0 (per half-wave), col = n16
#pragma unroll
    for (int nn = 0; nn < 4; ++nn) {
        int n = n0 + nn * 16 + n16;
        float bv = bias[n];
#pragma unroll
        for (int r = 0; r < 8; ++r) {
            int m = m0 + wave * 16 + r + b0;
            float v = acc[nn][r] + bv;
            if (mode == 0) {
                int b = m >> 11, l = m & 2047;  // m = b*L + l
                int h = n >> 7, d = n & 127;    // n = h*HD + d
                outB[((((size_t)b * NUM_HEADS + h) * SEQ_LEN) + l) * HEAD_DIM + d] =
                    (bf16)v;
            } else if (mode == 2) {
                int b = m >> 11, l = m & 2047;
                int h = n >> 7, d = n & 127;
                outB[((((size_t)b * NUM_HEADS + h) * HEAD_DIM) + d) * SEQ_LEN + l] =
                    (bf16)v;
            } else {
                outF[(size_t)m * N + n] = v;
            }
        }
    }
}

// ---------------------------------------------------------------------------
// Flash attention, transposed formulation, async double-buffered K/V tiles.
//   S^T = K Q^T   (A = K rows: contiguous; B = Q frags held in registers)
//   softmax: each lane owns one query column -> 1 shfl_xor(16) per reduction
//   O^T = V^T P^T (A = V^T rows: contiguous; B = P^T already in-lane)
// One workgroup per (head, 128 query rows); 32-key tiles staged in LDS.
// ---------------------------------------------------------------------------
__global__ __launch_bounds__(256)
void flash_attn_wmma(const bf16* __restrict__ Qb, const bf16* __restrict__ Kb,
                     const bf16* __restrict__ VbT, bf16* __restrict__ Ob) {
    const int t    = threadIdx.x;
    const int lane = t & 31;
    const int wave = t >> 5;
    const int bh   = blockIdx.y;            // 0..31
    const int b    = bh >> 4, h = bh & 15;
    const int lw   = blockIdx.x * 128 + wave * 16;
    const int b0   = (lane >= 16) ? 8 : 0;
    const int n16  = lane & 15;

    __shared__ bf16 Kt [2][ 32 * 136];  // [key][d], 272B row stride
    __shared__ bf16 VtT[2][128 *  40];  // [d][key], 80B row stride

    const size_t hoff = (size_t)bh * SEQ_LEN * HEAD_DIM;   // Q/K: [bh][l][d]
    // V^T layout [bh][d][l]: same per-head element count

    // per-thread staging coordinates
    const int kr = t >> 3, kc = (t & 7) * 16;   // K: 8 threads/row
    const int vr = t >> 1, vc = (t & 1) * 16;   // V^T: 2 threads/row
    const bf16* Kgl = Kb  + hoff + (size_t)kr * HEAD_DIM + kc;  // + j0*HEAD_DIM
    const bf16* Vgl = VbT + hoff + (size_t)vr * SEQ_LEN  + vc;  // + j0
    const int koff = kr * 136 + kc;
    const int voff = vr * 40 + vc;

    // Q fragments (B-operand layout), with softmax scale folded in
    bf16x16 q[4];
    {
        const bf16* qrow = Qb + hoff + (size_t)(lw + n16) * HEAD_DIM;
#pragma unroll
        for (int c = 0; c < 4; ++c)
#pragma unroll
            for (int e = 0; e < 16; ++e)
                q[c][e] = (bf16)((float)qrow[c * 32 + kmap(e, b0)] * ATTN_SCALE);
    }

    floatx8 o[8] = {};          // O^T: rows = d (8 subtiles), col = this query
    float mq = -1e30f, lq = 0.0f;   // per-lane scalars (one query per lane)

    // prologue: async-stage tile 0
    async_copy_b128(&Kt [0][koff],     Kgl);
    async_copy_b128(&Kt [0][koff + 8], Kgl + 8);
    async_copy_b128(&VtT[0][voff],     Vgl);
    async_copy_b128(&VtT[0][voff + 8], Vgl + 8);

    int pb = 0;
    for (int j0 = 0; j0 < SEQ_LEN; j0 += 32) {
        wait_async_all();
        __syncthreads();

        if (j0 + 32 < SEQ_LEN) {   // DMA next 32-key tile into alternate buffer
            int nb = pb ^ 1;
            const bf16* kg = Kgl + (size_t)(j0 + 32) * HEAD_DIM;
            const bf16* vg = Vgl + (j0 + 32);
            async_copy_b128(&Kt [nb][koff],     kg);
            async_copy_b128(&Kt [nb][koff + 8], kg + 8);
            async_copy_b128(&VtT[nb][voff],     vg);
            async_copy_b128(&VtT[nb][voff + 8], vg + 8);
        }

        // S^T = K Q^T : s0 = keys j0..j0+15, s1 = keys j0+16..j0+31
        floatx8 s0 = {}, s1 = {};
#pragma unroll
        for (int c = 0; c < 4; ++c) {
            bf16x16 a0, a1;
#pragma unroll
            for (int e = 0; e < 16; ++e) {
                int kk = c * 32 + kmap(e, b0);
                a0[e] = Kt[pb][n16 * 136 + kk];          // A rows = keys 0..15
                a1[e] = Kt[pb][(16 + n16) * 136 + kk];   // A rows = keys 16..31
            }
            s0 = wmma_bf16(a0, q[c], s0);
            s1 = wmma_bf16(a1, q[c], s1);
        }

        // online softmax: lane owns query n16; its 16 scores + partner's 16
        float mx = -1e30f;
#pragma unroll
        for (int r = 0; r < 8; ++r) {
            mx = fmaxf(mx, s0[r]);
            mx = fmaxf(mx, s1[r]);
        }
        mx = fmaxf(mx, __shfl_xor(mx, 16, 32));
        float newm  = fmaxf(mq, mx);
        float alpha = __expf(mq - newm);
        mq = newm;

        float sum = 0.0f;
        bf16x16 bp;   // P^T fragment: exactly the keys this lane holds
#pragma unroll
        for (int r = 0; r < 8; ++r) {
            float p0 = __expf(s0[r] - newm);
            float p1 = __expf(s1[r] - newm);
            sum += p0 + p1;
            bp[r]     = (bf16)p0;
            bp[r + 8] = (bf16)p1;
        }
        sum += __shfl_xor(sum, 16, 32);
        lq = lq * alpha + sum;

#pragma unroll
        for (int dt = 0; dt < 8; ++dt)
#pragma unroll
            for (int r = 0; r < 8; ++r) o[dt][r] *= alpha;

        // O^T += V^T P^T : A rows = d (contiguous in VtT), B = bp (registers)
#pragma unroll
        for (int dt = 0; dt < 8; ++dt) {
            bf16x16 av;
#pragma unroll
            for (int e = 0; e < 16; ++e)
                av[e] = VtT[pb][(dt * 16 + n16) * 40 + kmap(e, b0)];
            o[dt] = wmma_bf16(av, bp, o[dt]);
        }
        pb ^= 1;
    }

    // epilogue: lane writes its query's row; 8 consecutive d per acc -> b128
    float inv = 1.0f / lq;
    const int l = lw + n16;
    const size_t rowb = ((size_t)b * SEQ_LEN + l) * MODEL_DIM + (size_t)h * HEAD_DIM;
#pragma unroll
    for (int dt = 0; dt < 8; ++dt) {
        union { bf16 hh[8]; uint4 u; } pk;
#pragma unroll
        for (int r = 0; r < 8; ++r) pk.hh[r] = (bf16)(o[dt][r] * inv);
        *(uint4*)(Ob + rowb + dt * 16 + b0) = pk.u;
    }
}

// ---------------------------------------------------------------------------
// Host-side orchestration
// ---------------------------------------------------------------------------
extern "C" void kernel_launch(void* const* d_in, const int* in_sizes, int n_in,
                              void* d_out, int out_size, void* d_ws, size_t ws_size,
                              hipStream_t stream) {
    (void)in_sizes; (void)n_in; (void)out_size; (void)ws_size;

    const float* x_q = (const float*)d_in[0];
    const float* x_k = (const float*)d_in[1];
    const float* x_v = (const float*)d_in[2];
    const float* Wq  = (const float*)d_in[3];
    const float* bq  = (const float*)d_in[4];
    const float* Wk  = (const float*)d_in[5];
    const float* bk  = (const float*)d_in[6];
    const float* Wv  = (const float*)d_in[7];
    const float* bv  = (const float*)d_in[8];
    const float* Wo  = (const float*)d_in[9];
    const float* bo  = (const float*)d_in[10];
    float* out = (float*)d_out;

    const size_t XN = (size_t)M_TOTAL * MODEL_DIM;   // 8,388,608
    const size_t WN = (size_t)MODEL_DIM * MODEL_DIM; // 4,194,304

    char* ws = (char*)d_ws;
    size_t off = 0;
    bf16* xq_h  = (bf16*)(ws + off); off += XN * 2;
    bf16* xk_h  = (bf16*)(ws + off); off += XN * 2;
    bf16* xv_h  = (bf16*)(ws + off); off += XN * 2;
    bf16* wqT_h = (bf16*)(ws + off); off += WN * 2;
    bf16* wkT_h = (bf16*)(ws + off); off += WN * 2;
    bf16* wvT_h = (bf16*)(ws + off); off += WN * 2;
    bf16* woT_h = (bf16*)(ws + off); off += WN * 2;
    bf16* Qb    = (bf16*)(ws + off); off += XN * 2;
    bf16* Kb    = (bf16*)(ws + off); off += XN * 2;
    bf16* VbT   = (bf16*)(ws + off); off += XN * 2;  // [bh][d][l]
    bf16* Ab    = (bf16*)(ws + off); off += XN * 2;  // attn output, merged heads

    dim3 blk(256);

    // 1) one-time conversions: activations plain, weights transposed
    cvt_f32_to_bf16<<<(unsigned)(XN / 1024), blk, 0, stream>>>(x_q, xq_h, (int)XN);
    cvt_f32_to_bf16<<<(unsigned)(XN / 1024), blk, 0, stream>>>(x_k, xk_h, (int)XN);
    cvt_f32_to_bf16<<<(unsigned)(XN / 1024), blk, 0, stream>>>(x_v, xv_h, (int)XN);
    dim3 tgrid(MODEL_DIM / 32, MODEL_DIM / 32);      // (64, 64)
    cvt_transpose_f32_bf16<<<tgrid, blk, 0, stream>>>(Wq, wqT_h, MODEL_DIM);
    cvt_transpose_f32_bf16<<<tgrid, blk, 0, stream>>>(Wk, wkT_h, MODEL_DIM);
    cvt_transpose_f32_bf16<<<tgrid, blk, 0, stream>>>(Wv, wvT_h, MODEL_DIM);
    cvt_transpose_f32_bf16<<<tgrid, blk, 0, stream>>>(Wo, woT_h, MODEL_DIM);

    // 2) projections
    dim3 ggrid(MODEL_DIM / 64, M_TOTAL / 128);       // (32, 32)
    gemm_bf16_wmma<<<ggrid, blk, 0, stream>>>(xq_h, wqT_h, bq, Qb, nullptr,
                                              M_TOTAL, MODEL_DIM, MODEL_DIM, 0);
    gemm_bf16_wmma<<<ggrid, blk, 0, stream>>>(xk_h, wkT_h, bk, Kb, nullptr,
                                              M_TOTAL, MODEL_DIM, MODEL_DIM, 0);
    gemm_bf16_wmma<<<ggrid, blk, 0, stream>>>(xv_h, wvT_h, bv, VbT, nullptr,
                                              M_TOTAL, MODEL_DIM, MODEL_DIM, 2);

    // 3) flash attention per (head, 128-row query block)
    dim3 agrid(SEQ_LEN / 128, BATCH * NUM_HEADS);    // (16, 32)
    flash_attn_wmma<<<agrid, blk, 0, stream>>>(Qb, Kb, VbT, Ab);

    // 4) output projection -> f32 d_out
    gemm_bf16_wmma<<<ggrid, blk, 0, stream>>>(Ab, woT_h, bo, nullptr, out,
                                              M_TOTAL, MODEL_DIM, MODEL_DIM, 1);
}
